// Violin_Attention_20675972563545
// MI455X (gfx1250) — matrix operand order repaired and checked by reading, not compiled, not run on hardware
//
#include <hip/hip_runtime.h>
#include <cstdint>
#include <cstddef>

// ---------------- types ----------------
typedef __attribute__((ext_vector_type(16))) _Float16 v16h;
typedef __attribute__((ext_vector_type(8)))  _Float16 v8h;
typedef __attribute__((ext_vector_type(8)))  float    v8f;

// ---------------- problem constants ----------------
#define BB      128
#define NH      12
#define NN      197
#define NP      196
#define CC      768
#define HD      64
#define NCURVE  6
#define BN      (BB*NN)          // 25216 = 197*128 rows (multiple of 128!)
#define C3      (3*CC)           // 2304 (multiple of 128)
#define NPAD    224              // 14*16 attention K pad
#define QPAD    208              // 13*16 q row pad
#define SCALE   0.125f           // 64^-0.5
#define PLD     232              // padded LDS stride for P tile (halves)
#define TLD     40               // padded LDS stride for 32-K tiles (halves)

// ---------------- workspace layout (bytes, 256-aligned) ----------------
#define XH_OFF   ((size_t)0)                         // f16 [BN][CC]
#define WQ_OFF   (XH_OFF + (size_t)BN*CC*2)          // f16 [C3][CC]
#define WP_OFF   (WQ_OFF + (size_t)C3*CC*2)          // f16 [CC][CC]
#define QH_OFF   (WP_OFF + (size_t)CC*CC*2)          // f16 [B][H][QPAD][HD]
#define KH_OFF   (QH_OFF + (size_t)BB*NH*QPAD*HD*2)  // f16 [B][H][NPAD][HD]
#define VT_OFF   (KH_OFF + (size_t)BB*NH*NPAD*HD*2)  // f16 [B][H][HD][NPAD]
#define PO_OFF   (VT_OFF + (size_t)BB*NH*HD*NPAD*2)  // f16 [BN][CC]
#define MK_OFF   (PO_OFF + (size_t)BN*CC*2)          // f32 [H][N][N]

// ============ CDNA5 async copy global->LDS (ASYNCcnt path) ============
__device__ __forceinline__ uint32_t lds_off(const void* p) {
  // generic LDS pointer: addr[63:32]=aperture, addr[31:0]=LDS byte offset
  return (uint32_t)(uintptr_t)p;
}
__device__ __forceinline__ void async_b128(uint32_t ldsoff, const _Float16* g) {
  asm volatile("global_load_async_to_lds_b128 %0, %1, off"
               :: "v"(ldsoff), "v"(g) : "memory");
}
__device__ __forceinline__ void wait_async0() {
  asm volatile("s_wait_asynccnt 0" ::: "memory");
}

// ============ fragment loaders (ISA 7.12.2 layouts) ============
// A 16x32 f16: lane r=lane&15 is row M; element e -> K = 16*(e>>3) + 8*half + (e&7)
__device__ __forceinline__ v16h load_a_frag(const _Float16* base, int ld, int row0,
                                            int k0, int lane) {
  const int r = lane & 15, half = (lane >> 4) & 1;
  const _Float16* p = base + (size_t)(row0 + r) * ld + k0 + half * 8;
  v8h lo = *(const v8h*)p;
  v8h hi = *(const v8h*)(p + 16);
  v16h out;
#pragma unroll
  for (int e = 0; e < 8; ++e) { out[e] = lo[e]; out[e + 8] = hi[e]; }
  return out;
}
// B 32x16 f16 given as Bt[N][K] row-major: lane r = column N; e -> K = 16*(e>>3) + 8*half + (e&7)
__device__ __forceinline__ v16h load_b_frag(const _Float16* base, int ld, int row0,
                                            int k0, int lane) {
  const int r = lane & 15, half = (lane >> 4) & 1;
  const _Float16* p = base + (size_t)(row0 + r) * ld + k0 + half * 8;
  v8h lo = *(const v8h*)p;
  v8h hi = *(const v8h*)(p + 16);
  v16h out;
#pragma unroll
  for (int e = 0; e < 8; ++e) { out[e] = lo[e]; out[e + 8] = hi[e]; }
  return out;
}
__device__ __forceinline__ v8f wmma_f16(v16h a, v16h b, v8f c) {
  return __builtin_amdgcn_wmma_f32_16x16x32_f16(false, a, false, b, (short)0, c,
                                                false, false);
}

// ============ shared GEMM building blocks (128x128 block, 8 waves) ============
// LDS tile: 128 rows x 32 K-halves, padded row stride TLD=40 halves (80B) -> no
// bank conflicts (80 mod 256 spans all quads). One b128 chunk = (row, kc 0..3).
__device__ __forceinline__ void stage_tiles(const _Float16* A, const _Float16* B,
                                            int ldA, int ldB, int m0, int n0, int k0,
                                            uint32_t la_base, uint32_t lb_base,
                                            int tid) {
#pragma unroll
  for (int c = 0; c < 2; ++c) {
    const int ch = tid + c * 256;        // 0..511
    const int row = ch >> 2, kc = ch & 3;
    async_b128(la_base + row * (TLD * 2) + kc * 16,
               A + (size_t)(m0 + row) * ldA + k0 + kc * 8);
    async_b128(lb_base + row * (TLD * 2) + kc * 16,
               B + (size_t)(n0 + row) * ldB + k0 + kc * 8);
  }
}
__device__ __forceinline__ void compute_step(const _Float16* As, const _Float16* Bs,
                                             int wm, int wn, int lane,
                                             v8f acc[4][2]) {
  v16h af[4], bf[2];
#pragma unroll
  for (int mf = 0; mf < 4; ++mf) af[mf] = load_a_frag(As, TLD, wm * 64 + mf * 16, 0, lane);
#pragma unroll
  for (int nf = 0; nf < 2; ++nf) bf[nf] = load_b_frag(Bs, TLD, wn * 32 + nf * 16, 0, lane);
#pragma unroll
  for (int mf = 0; mf < 4; ++mf)
#pragma unroll
    for (int nf = 0; nf < 2; ++nf)
      acc[mf][nf] = wmma_f16(af[mf], bf[nf], acc[mf][nf]);
}

// ============ small utility kernels ============
__global__ void cvt_f32_to_f16(const float* __restrict__ in,
                               _Float16* __restrict__ out, int n) {
  for (int i = blockIdx.x * blockDim.x + threadIdx.x; i < n;
       i += gridDim.x * blockDim.x)
    out[i] = (_Float16)in[i];
}
__global__ void zero_fill_u32(uint32_t* __restrict__ p, int n) {
  for (int i = blockIdx.x * blockDim.x + threadIdx.x; i < n;
       i += gridDim.x * blockDim.x)
    p[i] = 0u;
}

// ============ decay mask M[h,i,j] ============
__global__ void mask_kernel(const float* __restrict__ ai,
                            const int* __restrict__ cidx,
                            float* __restrict__ M) {
  const int j = threadIdx.x;
  const int i = blockIdx.x % NN;
  const int h = blockIdx.x / NN;
  if (j >= NN) return;
  float val;
  if (i == 0 || j == 0) {
    val = 1.0f;
  } else {
    float acc = 0.0f;
#pragma unroll
    for (int c = 0; c < NCURVE; ++c) {
      const float a   = ai[c * NH + h];
      const float p   = 1.0f / (1.0f + __expf(-a));
      const float l2p = __log2f(p);
      const int ii = cidx[c * NP + (i - 1)];
      const int jj = cidx[c * NP + (j - 1)];
      acc += exp2f(l2p * fabsf((float)(ii - jj)));     // p^|di|
    }
    val = acc * (1.0f / (float)NCURVE);
  }
  M[((size_t)h * NN + i) * NN + j] = val;
}

// ============ QKV GEMM: [BN x CC] x [C3 x CC]^T, scatter to q/k/vT ============
__global__ __launch_bounds__(256) void qkv_gemm(const _Float16* __restrict__ xh,
                                                const _Float16* __restrict__ wq,
                                                _Float16* __restrict__ qh,
                                                _Float16* __restrict__ kh,
                                                _Float16* __restrict__ vT) {
  __shared__ _Float16 ldsA[2 * 128 * TLD];   // 20.5 KB
  __shared__ _Float16 ldsB[2 * 128 * TLD];

  const int tid = threadIdx.x, lane = tid & 31, w = tid >> 5;
  const int wm = w >> 2, wn = w & 3;
  const int n0 = blockIdx.x * 128;           // over C3
  const int m0 = blockIdx.y * 128;           // over BN
  const uint32_t laB = lds_off(ldsA), lbB = lds_off(ldsB);

  v8f acc[4][2] = {};
  stage_tiles(xh, wq, CC, CC, m0, n0, 0, laB, lbB, tid);
  for (int kt = 0; kt < CC / 32; ++kt) {
    wait_async0();
    __syncthreads();
    const int buf = kt & 1;
    if (kt + 1 < CC / 32)
      stage_tiles(xh, wq, CC, CC, m0, n0, (kt + 1) * 32,
                  laB + (buf ^ 1) * 128 * TLD * 2, lbB + (buf ^ 1) * 128 * TLD * 2, tid);
    compute_step(ldsA + buf * 128 * TLD, ldsB + buf * 128 * TLD, wm, wn, lane, acc);
    __syncthreads();
  }

  const int half = (lane >> 4) & 1;
#pragma unroll
  for (int mf = 0; mf < 4; ++mf) {
#pragma unroll
    for (int g = 0; g < 8; ++g) {
      const int m = m0 + wm * 64 + mf * 16 + g + 8 * half;  // token row
      const int b = m / NN;
      const int i = m - b * NN;
#pragma unroll
      for (int nf = 0; nf < 2; ++nf) {
        const int colb = n0 + wn * 32 + nf * 16;            // tile-uniform base
        const int t    = colb / CC;                         // 0:q 1:k 2:v
        const int h    = (colb % CC) / HD;
        const int d    = (colb % HD) + (lane & 15);
        const size_t bh = (size_t)b * NH + h;
        const float v = acc[mf][nf][g];
        if (t == 0)      qh[(bh * QPAD + i) * HD + d] = (_Float16)(v * SCALE);
        else if (t == 1) kh[(bh * NPAD + i) * HD + d] = (_Float16)v;
        else             vT[(bh * HD + d) * NPAD + i] = (_Float16)v;
      }
    }
  }
}

// ============ fused attention: one wave per (b, h, 16-row q tile) ============
__global__ __launch_bounds__(32) void attn_kernel(const _Float16* __restrict__ qh,
                                                  const _Float16* __restrict__ kh,
                                                  const _Float16* __restrict__ vTh,
                                                  const float* __restrict__ M,
                                                  _Float16* __restrict__ Pout) {
  __shared__ _Float16 Plds[16 * PLD];  // padded stride: conflict-free TR bounce

  const int lane = threadIdx.x;
  const int qt = blockIdx.x % (QPAD / 16);
  const int bh = blockIdx.x / (QPAD / 16);
  const int h  = bh % NH;
  const int b  = bh / NH;

  const _Float16* q  = qh  + (size_t)bh * QPAD * HD;
  const _Float16* k  = kh  + (size_t)bh * NPAD * HD;
  const _Float16* vT = vTh + (size_t)bh * HD * NPAD;

  const int half = (lane >> 4) & 1;
  const int col0 = lane & 15;

  // hoisted q fragments (K = 64 -> 2)
  v16h aq[2];
  aq[0] = load_a_frag(q, HD, qt * 16, 0, lane);
  aq[1] = load_a_frag(q, HD, qt * 16, 32, lane);

  float s[NPAD / 16][8];
#pragma unroll
  for (int jt = 0; jt < NPAD / 16; ++jt) {
    v8f acc = {};
#pragma unroll
    for (int kk = 0; kk < 2; ++kk)
      acc = wmma_f16(aq[kk], load_b_frag(k, HD, jt * 16, kk * 32, lane), acc);
    const int col = jt * 16 + col0;
#pragma unroll
    for (int g = 0; g < 8; ++g) {
      const int row = qt * 16 + g + 8 * half;
      if (row < NN && col < NN)
        s[jt][g] = acc[g] * M[((size_t)h * NN + row) * NN + col];
      else
        s[jt][g] = -__builtin_inff();
    }
  }

  // row softmax on 16-lane halves
#pragma unroll
  for (int g = 0; g < 8; ++g) {
    float m = -__builtin_inff();
#pragma unroll
    for (int jt = 0; jt < NPAD / 16; ++jt) m = fmaxf(m, s[jt][g]);
#pragma unroll
    for (int d = 1; d < 16; d <<= 1) m = fmaxf(m, __shfl_xor(m, d, 32));
    const float mm = (m == -__builtin_inff()) ? 0.0f : m;
    float sum = 0.0f;
#pragma unroll
    for (int jt = 0; jt < NPAD / 16; ++jt) {
      const float e = __expf(s[jt][g] - mm);
      s[jt][g] = e;
      sum += e;
    }
#pragma unroll
    for (int d = 1; d < 16; d <<= 1) sum += __shfl_xor(sum, d, 32);
    const float inv = 1.0f / fmaxf(sum, 1e-20f);
    const int row_l = g + 8 * half;
#pragma unroll
    for (int jt = 0; jt < NPAD / 16; ++jt)
      Plds[row_l * PLD + jt * 16 + col0] = (_Float16)(s[jt][g] * inv);
  }
  __syncthreads();

  // cache P fragments once (A-layout), reuse across all 4 output tiles
  v16h pf[NPAD / 32];
#pragma unroll
  for (int kk = 0; kk < NPAD / 32; ++kk)
    pf[kk] = load_a_frag(Plds, PLD, 0, kk * 32, lane);

#pragma unroll
  for (int ntile = 0; ntile < HD / 16; ++ntile) {
    v8f acc = {};
#pragma unroll
    for (int kk = 0; kk < NPAD / 32; ++kk)
      acc = wmma_f16(pf[kk], load_b_frag(vT, NPAD, ntile * 16, kk * 32, lane), acc);
    const int col = h * HD + ntile * 16 + col0;
#pragma unroll
    for (int g = 0; g < 8; ++g) {
      const int row = qt * 16 + g + 8 * half;
      if (row < NN)
        Pout[((size_t)b * NN + row) * CC + col] = (_Float16)acc[g];
    }
  }
}

// ============ projection: [BN x CC] x [CC x CC]^T + bias -> f32 ============
__global__ __launch_bounds__(256) void proj_gemm(const _Float16* __restrict__ ph,
                                                 const _Float16* __restrict__ wp,
                                                 const float* __restrict__ bias,
                                                 float* __restrict__ out) {
  __shared__ _Float16 ldsA[2 * 128 * TLD];
  __shared__ _Float16 ldsB[2 * 128 * TLD];

  const int tid = threadIdx.x, lane = tid & 31, w = tid >> 5;
  const int wm = w >> 2, wn = w & 3;
  const int n0 = blockIdx.x * 128;           // over CC
  const int m0 = blockIdx.y * 128;           // over BN
  const uint32_t laB = lds_off(ldsA), lbB = lds_off(ldsB);

  v8f acc[4][2] = {};
  stage_tiles(ph, wp, CC, CC, m0, n0, 0, laB, lbB, tid);
  for (int kt = 0; kt < CC / 32; ++kt) {
    wait_async0();
    __syncthreads();
    const int buf = kt & 1;
    if (kt + 1 < CC / 32)
      stage_tiles(ph, wp, CC, CC, m0, n0, (kt + 1) * 32,
                  laB + (buf ^ 1) * 128 * TLD * 2, lbB + (buf ^ 1) * 128 * TLD * 2, tid);
    compute_step(ldsA + buf * 128 * TLD, ldsB + buf * 128 * TLD, wm, wn, lane, acc);
    __syncthreads();
  }

  const int half = (lane >> 4) & 1;
#pragma unroll
  for (int nf = 0; nf < 2; ++nf) {
    const int col = n0 + wn * 32 + nf * 16 + (lane & 15);
    const float bv = bias[col];
#pragma unroll
    for (int mf = 0; mf < 4; ++mf)
#pragma unroll
      for (int g = 0; g < 8; ++g) {
        const int m = m0 + wm * 64 + mf * 16 + g + 8 * half;
        out[(size_t)m * CC + col] = acc[mf][nf][g] + bv;
      }
  }
}

// ============ launcher ============
extern "C" void kernel_launch(void* const* d_in, const int* in_sizes, int n_in,
                              void* d_out, int out_size, void* d_ws, size_t ws_size,
                              hipStream_t stream) {
  const float* x      = (const float*)d_in[0];
  const float* w_qkv  = (const float*)d_in[1];
  const float* w_proj = (const float*)d_in[2];
  const float* b_proj = (const float*)d_in[3];
  const float* ai     = (const float*)d_in[4];
  const int*   cidx   = (const int*)d_in[5];
  float* out = (float*)d_out;

  char* ws = (char*)d_ws;
  _Float16* xh  = (_Float16*)(ws + XH_OFF);
  _Float16* wqh = (_Float16*)(ws + WQ_OFF);
  _Float16* wph = (_Float16*)(ws + WP_OFF);
  _Float16* qh  = (_Float16*)(ws + QH_OFF);
  _Float16* kh  = (_Float16*)(ws + KH_OFF);
  _Float16* vT  = (_Float16*)(ws + VT_OFF);
  _Float16* po  = (_Float16*)(ws + PO_OFF);
  float*    M   = (float*)(ws + MK_OFF);

  cvt_f32_to_f16<<<2048, 256, 0, stream>>>(x, xh, BN * CC);
  cvt_f32_to_f16<<<512, 256, 0, stream>>>(w_qkv, wqh, C3 * CC);
  cvt_f32_to_f16<<<512, 256, 0, stream>>>(w_proj, wph, CC * CC);

  {  // zero padded q/k/vT region so attention needs no load guards
    const int nwords = (int)(((size_t)BB * NH * QPAD * HD +
                              (size_t)BB * NH * NPAD * HD * 2) * 2 / 4);
    zero_fill_u32<<<2048, 256, 0, stream>>>((uint32_t*)(ws + QH_OFF), nwords);
  }

  mask_kernel<<<NH * NN, 256, 0, stream>>>(ai, cidx, M);

  {  // QKV: grid = (C3/128, BN/128) = (18, 197)
    dim3 grid(C3 / 128, BN / 128);
    qkv_gemm<<<grid, 256, 0, stream>>>(xh, wqh, qh, kh, vT);
  }

  attn_kernel<<<BB * NH * (QPAD / 16), 32, 0, stream>>>(qh, kh, vT, M, po);

  {  // proj: grid = (CC/128, BN/128) = (6, 197)
    dim3 grid(CC / 128, BN / 128);
    proj_gemm<<<grid, 256, 0, stream>>>(po, wph, b_proj, out);
  }
}
